// LinearAttention_45268955300512
// MI455X (gfx1250) — compile-verified, matrix operand order checked by reading, not run on hardware
//
#include <hip/hip_runtime.h>
#include <hip/hip_bf16.h>

// ---------------------------------------------------------------------------
// Linear attention (B=4, T=4096, C=2048, H=16, Dh=128) for MI455X / gfx1250.
// All contractions on v_wmma_f32_16x16x32_bf16 (wave32 WMMA). Tiles are moved
// with the CDNA5 async DMA path (global_load_async_to_lds_b128 + ASYNCcnt),
// double-buffered in LDS; fragments come from ds_load_b128.
// ---------------------------------------------------------------------------

typedef __bf16 bf16;
typedef __attribute__((ext_vector_type(16))) __bf16 v16bf;
typedef __attribute__((ext_vector_type(8)))  __bf16 v8bf;
typedef __attribute__((ext_vector_type(4)))  __bf16 v4bf;
typedef __attribute__((ext_vector_type(8)))  float  v8f;

constexpr int Bb = 4;
constexpr int T  = 4096;
constexpr int C  = 2048;
constexpr int H  = 16;
constexpr int Dh = 128;
constexpr int BT = Bb * T;           // 16384 tokens
constexpr int KVSPLIT = 8;           // T-split for the per-head kv reduction
constexpr int TCHUNK  = T / KVSPLIT; // 512
constexpr unsigned TILE_BYTES = 128 * 32 * 2;  // one 128x32 bf16 tile = 8 KB

// ---------------------------------------------------------------------------
// Fragment load from a row-major [128 x 32] bf16 LDS tile.
// Lane layout per ISA: rows = (lane&15), K-chunks (lane<16 ? 0 : 8) and +16.
// ---------------------------------------------------------------------------
__device__ __forceinline__ v16bf ldfrag(const bf16* base, int row0, int lane) {
  const bf16* p = base + (row0 + (lane & 15)) * 32 + ((lane >> 4) << 3);
  v8bf lo = *(const v8bf*)p;          // ds_load_b128
  v8bf hi = *(const v8bf*)(p + 16);   // ds_load_b128
  v16bf r;
#pragma unroll
  for (int i = 0; i < 8; ++i) { r[i] = lo[i]; r[i + 8] = hi[i]; }
  return r;
}

// ---------------------------------------------------------------------------
// Async DMA of a 128x32 bf16 tile (g at tile origin, row stride ld elements)
// straight into LDS: each of the 256 threads issues two async b128 copies.
// Tracked by ASYNCcnt; consumer does s_wait_asynccnt 0 + barrier.
// ---------------------------------------------------------------------------
__device__ __forceinline__ void async_tile(unsigned ldsT, const bf16* g,
                                           int ld, int tid) {
  int r = tid >> 1;
  int c = (tid & 1) << 4;
  const bf16* gp = g + (size_t)r * ld + c;
  unsigned lp = ldsT + (unsigned)(r * 32 + c) * 2u;
  asm volatile("global_load_async_to_lds_b128 %0, %1, off"
               :: "v"(lp), "v"(gp) : "memory");
  asm volatile("global_load_async_to_lds_b128 %0, %1, off offset:16"
               :: "v"(lp), "v"(gp) : "memory");
}

__device__ __forceinline__ void wait_async0() {
  asm volatile("s_wait_asynccnt 0x0" ::: "memory");
}

// LDS offset (addr[31:0] of the flat aperture address) for async destinations.
__device__ __forceinline__ unsigned lds_off(const void* p) {
  return (unsigned)(size_t)p;
}

// One 32-deep K step: fragments into distinct registers, then 8 WMMAs.
__device__ __forceinline__ void mma_tile(const bf16* sA, const bf16* sB,
                                         int wm, int wn, int lane, v8f acc[2][4]) {
  v16bf a[2], b[4];
#pragma unroll
  for (int i = 0; i < 2; ++i) a[i] = ldfrag(sA, wm * 32 + i * 16, lane);
#pragma unroll
  for (int j = 0; j < 4; ++j) b[j] = ldfrag(sB, wn * 64 + j * 16, lane);
#pragma unroll
  for (int j = 0; j < 4; ++j)
#pragma unroll
    for (int i = 0; i < 2; ++i)
      acc[i][j] = __builtin_amdgcn_wmma_f32_16x16x32_bf16(
          false, a[i], false, b[j], (short)0, acc[i][j], false, false);
}

__device__ __forceinline__ void zero_acc(v8f acc[2][4]) {
  const v8f z = {0.f, 0.f, 0.f, 0.f, 0.f, 0.f, 0.f, 0.f};
#pragma unroll
  for (int i = 0; i < 2; ++i)
#pragma unroll
    for (int j = 0; j < 4; ++j) acc[i][j] = z;
}

__device__ __forceinline__ float feat(float x) {      // elu(x)+1
  return x > 0.f ? x + 1.f : __expf(x);
}

// ---------------------------------------------------------------------------
// K0: fp32 -> bf16 conversion (vectorized 4 elems/thread).
// ---------------------------------------------------------------------------
__global__ void cvt_f32_bf16(const float* __restrict__ src,
                             bf16* __restrict__ dst, int n) {
  int i = (blockIdx.x * blockDim.x + threadIdx.x) << 2;
  if (i >= n) return;
  float4 f = *(const float4*)(src + i);
  v4bf o;
  o[0] = (bf16)f.x; o[1] = (bf16)f.y; o[2] = (bf16)f.z; o[3] = (bf16)f.w;
  *(v4bf*)(dst + i) = o;
}

// ---------------------------------------------------------------------------
// K2: fused QKV projection. y = x @ W^T (M=BT, N=C, K=C), mode selects W and
// the epilogue: q -> feat, [B,H,T,Dh]; k -> feat, transposed [B,H,Dh,T];
// v -> transposed [B,H,Dh,T]. Async-DMA double-buffered LDS pipeline.
// ---------------------------------------------------------------------------
__global__ __launch_bounds__(256) void qkv_gemm(
    const bf16* __restrict__ xb,
    const bf16* __restrict__ Wq, const bf16* __restrict__ Wk,
    const bf16* __restrict__ Wv,
    bf16* __restrict__ qb, bf16* __restrict__ ktb, bf16* __restrict__ vtb) {
  __shared__ bf16 sA[2][128 * 32];
  __shared__ bf16 sB[2][128 * 32];
  int tid = threadIdx.x, lane = tid & 31, wave = tid >> 5;
  int wm = wave & 3, wn = wave >> 2;
  int m_base = blockIdx.y * 128;     // token rows
  int n_base = blockIdx.x * 128;     // output channels
  int mode = blockIdx.z;
  const bf16* W = (mode == 0) ? Wq : (mode == 1) ? Wk : Wv;
  const bf16* gA = xb + (size_t)m_base * C;
  const bf16* gB = W  + (size_t)n_base * C;
  unsigned aLds = lds_off(&sA[0][0]);
  unsigned bLds = lds_off(&sB[0][0]);

  v8f acc[2][4];
  zero_acc(acc);

  async_tile(aLds, gA, C, tid);
  async_tile(bLds, gB, C, tid);
  wait_async0();
  __syncthreads();

  int buf = 0;
  for (int k0 = 0; k0 < C; k0 += 32, buf ^= 1) {
    bool more = (k0 + 32) < C;
    if (more) {                       // async DMA of next tile into buf^1
      async_tile(aLds + (buf ^ 1) * TILE_BYTES, gA + k0 + 32, C, tid);
      async_tile(bLds + (buf ^ 1) * TILE_BYTES, gB + k0 + 32, C, tid);
    }
    mma_tile(sA[buf], sB[buf], wm, wn, lane, acc);
    if (more) wait_async0();          // DMA hidden behind the 8 WMMAs
    __syncthreads();
  }

#pragma unroll
  for (int i = 0; i < 2; ++i)
#pragma unroll
    for (int j = 0; j < 4; ++j)
#pragma unroll
      for (int r = 0; r < 8; ++r) {
        int gm = m_base + wm * 32 + i * 16 + r + ((lane >> 4) << 3);
        int gn = n_base + wn * 64 + j * 16 + (lane & 15);
        float v = acc[i][j][r];
        int b = gm >> 12, t = gm & (T - 1);
        int h = gn >> 7,  d = gn & (Dh - 1);
        if (mode == 2) {
          vtb[((size_t)(b * H + h) * Dh + d) * T + t] = (bf16)v;
        } else {
          float f = feat(v);
          if (mode == 0)
            qb[((size_t)(b * H + h) * T + t) * Dh + d] = (bf16)f;
          else
            ktb[((size_t)(b * H + h) * Dh + d) * T + t] = (bf16)f;
        }
      }
}

// ---------------------------------------------------------------------------
// K3a: per-head partial kv = k^T v over a T-chunk (M=Dh, N=Dh, K=TCHUNK),
// plus partial ksum. 512 blocks = (b,h) x KVSPLIT for full-device utilization.
// Partial output fp32, layout [bh*KVSPLIT+s][e*Dh+d] (already transposed).
// ---------------------------------------------------------------------------
__global__ __launch_bounds__(256) void kv_partial(
    const bf16* __restrict__ ktb, const bf16* __restrict__ vtb,
    float* __restrict__ kvpart, float* __restrict__ kspart) {
  __shared__ bf16 sA[2][128 * 32];
  __shared__ bf16 sB[2][128 * 32];
  int tid = threadIdx.x, lane = tid & 31, wave = tid >> 5;
  int wm = wave & 3, wn = wave >> 2;
  int s  = blockIdx.x & (KVSPLIT - 1);
  int bh = blockIdx.x >> 3;
  const bf16* Kt = ktb + (size_t)bh * Dh * T + s * TCHUNK;
  const bf16* Vt = vtb + (size_t)bh * Dh * T + s * TCHUNK;
  unsigned aLds = lds_off(&sA[0][0]);
  unsigned bLds = lds_off(&sB[0][0]);

  v8f acc[2][4];
  zero_acc(acc);
  float ks = 0.f;

  async_tile(aLds, Kt, T, tid);
  async_tile(bLds, Vt, T, tid);
  wait_async0();
  __syncthreads();

  int buf = 0;
  for (int t0 = 0; t0 < TCHUNK; t0 += 32, buf ^= 1) {
    bool more = (t0 + 32) < TCHUNK;
    if (more) {
      async_tile(aLds + (buf ^ 1) * TILE_BYTES, Kt + t0 + 32, T, tid);
      async_tile(bLds + (buf ^ 1) * TILE_BYTES, Vt + t0 + 32, T, tid);
    }
    if (tid < 128) {                   // fold ksum out of the staged k tile
#pragma unroll
      for (int c = 0; c < 32; ++c) ks += (float)sA[buf][tid * 32 + c];
    }
    mma_tile(sA[buf], sB[buf], wm, wn, lane, acc);
    if (more) wait_async0();
    __syncthreads();
  }

  float* out = kvpart + (size_t)blockIdx.x * Dh * Dh;
#pragma unroll
  for (int i = 0; i < 2; ++i)
#pragma unroll
    for (int j = 0; j < 4; ++j)
#pragma unroll
      for (int r = 0; r < 8; ++r) {
        int d = wm * 32 + i * 16 + r + ((lane >> 4) << 3);
        int e = wn * 64 + j * 16 + (lane & 15);
        out[e * Dh + d] = acc[i][j][r];
      }
  if (tid < 128) kspart[blockIdx.x * Dh + tid] = ks;
}

// ---------------------------------------------------------------------------
// K3b: deterministic reduce of the KVSPLIT partials -> kvT (bf16) + ksum.
// ---------------------------------------------------------------------------
__global__ __launch_bounds__(256) void kv_reduce(
    const float* __restrict__ kvpart, const float* __restrict__ kspart,
    bf16* __restrict__ kvT, float* __restrict__ ksum) {
  int tid = threadIdx.x;
  int bh = blockIdx.x;
  const float* base = kvpart + (size_t)bh * KVSPLIT * Dh * Dh;
  for (int i = tid; i < Dh * Dh; i += 256) {
    float sum = 0.f;
#pragma unroll
    for (int p = 0; p < KVSPLIT; ++p) sum += base[(size_t)p * Dh * Dh + i];
    kvT[(size_t)bh * Dh * Dh + i] = (bf16)sum;
  }
  if (tid < Dh) {
    float sum = 0.f;
#pragma unroll
    for (int p = 0; p < KVSPLIT; ++p)
      sum += kspart[(bh * KVSPLIT + p) * Dh + tid];
    ksum[bh * Dh + tid] = sum;
  }
}

// ---------------------------------------------------------------------------
// K4: out_h = (q @ kv) * z, z = 1/(q . ksum + 1e-6); writes attn[B,T,C] bf16.
// M = 128 tokens per block, N = Dh, K = Dh.
// ---------------------------------------------------------------------------
__global__ __launch_bounds__(256) void attn_out_gemm(
    const bf16* __restrict__ qb, const bf16* __restrict__ kvT,
    const float* __restrict__ ksum, bf16* __restrict__ attn) {
  __shared__ bf16 sA[2][128 * 32];
  __shared__ bf16 sB[2][128 * 32];
  __shared__ float sZ[128];
  int tid = threadIdx.x, lane = tid & 31, wave = tid >> 5;
  int wm = wave & 3, wn = wave >> 2;
  int tt = blockIdx.x & (T / 128 - 1);
  int bh = blockIdx.x >> 5;
  int b = bh >> 4, h = bh & (H - 1);
  const bf16* Q  = qb  + (size_t)bh * T * Dh + (size_t)(tt * 128) * Dh;
  const bf16* KV = kvT + (size_t)bh * Dh * Dh;
  unsigned aLds = lds_off(&sA[0][0]);
  unsigned bLds = lds_off(&sB[0][0]);

  async_tile(aLds, Q, Dh, tid);
  async_tile(bLds, KV, Dh, tid);

  if (tid < 128) {                       // per-token normalizer
    const bf16* qrow = Q + (size_t)tid * Dh;
    const float* kss = ksum + bh * Dh;
    float dp = 0.f;
#pragma unroll 8
    for (int d = 0; d < Dh; ++d) dp += (float)qrow[d] * kss[d];
    sZ[tid] = 1.f / (dp + 1e-6f);
  }

  v8f acc[2][4];
  zero_acc(acc);

  wait_async0();
  __syncthreads();

  int buf = 0;
  for (int k0 = 0; k0 < Dh; k0 += 32, buf ^= 1) {
    bool more = (k0 + 32) < Dh;
    if (more) {
      async_tile(aLds + (buf ^ 1) * TILE_BYTES, Q + k0 + 32, Dh, tid);
      async_tile(bLds + (buf ^ 1) * TILE_BYTES, KV + k0 + 32, Dh, tid);
    }
    mma_tile(sA[buf], sB[buf], wm, wn, lane, acc);
    if (more) wait_async0();
    __syncthreads();
  }

#pragma unroll
  for (int i = 0; i < 2; ++i)
#pragma unroll
    for (int j = 0; j < 4; ++j)
#pragma unroll
      for (int r = 0; r < 8; ++r) {
        int lm = wm * 32 + i * 16 + r + ((lane >> 4) << 3);
        int e  = wn * 64 + j * 16 + (lane & 15);
        float v = acc[i][j][r] * sZ[lm];
        int t = tt * 128 + lm;
        attn[((size_t)(b * T + t)) * C + h * Dh + e] = (bf16)v;
      }
}

// ---------------------------------------------------------------------------
// K5: final projection: out = attn @ Wo^T, fp32 output.
// ---------------------------------------------------------------------------
__global__ __launch_bounds__(256) void out_gemm(
    const bf16* __restrict__ attn, const bf16* __restrict__ Wo,
    float* __restrict__ out) {
  __shared__ bf16 sA[2][128 * 32];
  __shared__ bf16 sB[2][128 * 32];
  int tid = threadIdx.x, lane = tid & 31, wave = tid >> 5;
  int wm = wave & 3, wn = wave >> 2;
  int m_base = blockIdx.y * 128;
  int n_base = blockIdx.x * 128;
  const bf16* gA = attn + (size_t)m_base * C;
  const bf16* gB = Wo   + (size_t)n_base * C;
  unsigned aLds = lds_off(&sA[0][0]);
  unsigned bLds = lds_off(&sB[0][0]);

  v8f acc[2][4];
  zero_acc(acc);

  async_tile(aLds, gA, C, tid);
  async_tile(bLds, gB, C, tid);
  wait_async0();
  __syncthreads();

  int buf = 0;
  for (int k0 = 0; k0 < C; k0 += 32, buf ^= 1) {
    bool more = (k0 + 32) < C;
    if (more) {
      async_tile(aLds + (buf ^ 1) * TILE_BYTES, gA + k0 + 32, C, tid);
      async_tile(bLds + (buf ^ 1) * TILE_BYTES, gB + k0 + 32, C, tid);
    }
    mma_tile(sA[buf], sB[buf], wm, wn, lane, acc);
    if (more) wait_async0();
    __syncthreads();
  }

#pragma unroll
  for (int i = 0; i < 2; ++i)
#pragma unroll
    for (int j = 0; j < 4; ++j)
#pragma unroll
      for (int r = 0; r < 8; ++r) {
        int gm = m_base + wm * 32 + i * 16 + r + ((lane >> 4) << 3);
        int gn = n_base + wn * 64 + j * 16 + (lane & 15);
        out[(size_t)gm * C + gn] = acc[i][j][r];
      }
}

// ---------------------------------------------------------------------------
// Host-side launcher.
// ---------------------------------------------------------------------------
extern "C" void kernel_launch(void* const* d_in, const int* in_sizes, int n_in,
                              void* d_out, int out_size, void* d_ws, size_t ws_size,
                              hipStream_t stream) {
  const float* x  = (const float*)d_in[0];
  // d_in[1]=cos, d_in[2]=sin are unused by the reference module.
  const float* Wq = (const float*)d_in[3];
  const float* Wk = (const float*)d_in[4];
  const float* Wv = (const float*)d_in[5];
  const float* Wo = (const float*)d_in[6];

  // Carve scratch out of d_ws (256B-aligned slabs).
  size_t off = 0;
  auto carve = [&](size_t bytes) -> void* {
    void* p = (char*)d_ws + off;
    off += (bytes + 255) & ~(size_t)255;
    return p;
  };
  bf16* xb    = (bf16*)carve((size_t)BT * C * sizeof(bf16));          // 67 MB
  bf16* wqb   = (bf16*)carve((size_t)C * C * sizeof(bf16));           // 8.4 MB
  bf16* wkb   = (bf16*)carve((size_t)C * C * sizeof(bf16));
  bf16* wvb   = (bf16*)carve((size_t)C * C * sizeof(bf16));
  bf16* wob   = (bf16*)carve((size_t)C * C * sizeof(bf16));
  bf16* qb    = (bf16*)carve((size_t)Bb * H * T * Dh * sizeof(bf16)); // 67 MB
  bf16* ktb   = (bf16*)carve((size_t)Bb * H * Dh * T * sizeof(bf16)); // 67 MB
  bf16* vtb   = (bf16*)carve((size_t)Bb * H * Dh * T * sizeof(bf16)); // 67 MB
  float* kvpart = (float*)carve((size_t)Bb * H * KVSPLIT * Dh * Dh * sizeof(float)); // 33.5 MB
  float* kspart = (float*)carve((size_t)Bb * H * KVSPLIT * Dh * sizeof(float));
  float* ksum   = (float*)carve((size_t)Bb * H * Dh * sizeof(float));
  bf16* kvT   = (bf16*)carve((size_t)Bb * H * Dh * Dh * sizeof(bf16)); // 2 MB
  bf16* attn  = (bf16*)carve((size_t)BT * C * sizeof(bf16));           // 67 MB
  (void)ws_size; (void)in_sizes; (void)n_in; (void)out_size;

  // K0/K1: precision downcast (one pass; weights reused by heavy GEMMs).
  cvt_f32_bf16<<<(BT * C / 4 + 255) / 256, 256, 0, stream>>>(x, xb, BT * C);
  cvt_f32_bf16<<<(C * C / 4 + 255) / 256, 256, 0, stream>>>(Wq, wqb, C * C);
  cvt_f32_bf16<<<(C * C / 4 + 255) / 256, 256, 0, stream>>>(Wk, wkb, C * C);
  cvt_f32_bf16<<<(C * C / 4 + 255) / 256, 256, 0, stream>>>(Wv, wvb, C * C);
  cvt_f32_bf16<<<(C * C / 4 + 255) / 256, 256, 0, stream>>>(Wo, wob, C * C);

  // K2: fused QKV projection (3 GEMMs over grid.z), feature map + transposes.
  qkv_gemm<<<dim3(C / 128, BT / 128, 3), 256, 0, stream>>>(
      xb, wqb, wkb, wvb, qb, ktb, vtb);

  // K3: per-head kv summary + k-sum, split over T for full-device utilization,
  // then a deterministic reduction.
  kv_partial<<<Bb * H * KVSPLIT, 256, 0, stream>>>(ktb, vtb, kvpart, kspart);
  kv_reduce<<<Bb * H, 256, 0, stream>>>(kvpart, kspart, kvT, ksum);

  // K4: normalized q @ kv -> attn[B,T,C] (bf16).
  attn_out_gemm<<<Bb * H * (T / 128), 256, 0, stream>>>(qb, kvT, ksum, attn);

  // K5: output projection -> fp32 result.
  out_gemm<<<dim3(C / 128, BT / 128), 256, 0, stream>>>(attn, wob, (float*)d_out);
}
